// AttentionNetwork_15685220565777
// MI455X (gfx1250) — matrix operand order repaired and checked by reading, not compiled, not run on hardware
//
#include <hip/hip_runtime.h>
#include <hip/hip_bf16.h>

typedef __attribute__((ext_vector_type(16))) __bf16 v16bf;
typedef __attribute__((ext_vector_type(8)))  float  v8f;

#define P_    4096
#define LMAX_ 64
#define D_    512
#define H_    512
#define XPAD  516   // LDS row stride (floats) to break bank conflicts

// ---------------------------------------------------------------------------
// CDNA5 async global->LDS copy (16B per lane), tracked by ASYNCcnt.
// LDS destination = low 32 bits of the generic shared-memory pointer
// (flat LDS aperture: addr[31:0] is the LDS offset).
// ---------------------------------------------------------------------------
static __device__ __forceinline__ void async_copy16(void* lds_dst, const void* gsrc) {
  unsigned lds = (unsigned)(uintptr_t)lds_dst;
  asm volatile("global_load_async_to_lds_b128 %0, %1, off"
               :: "v"(lds), "v"(gsrc) : "memory");
}
static __device__ __forceinline__ void async_wait0() {
  asm volatile("s_wait_asynccnt 0" ::: "memory");
}

// ---------------------------------------------------------------------------
// Repack a [D,H] f32 weight matrix into bf16 WMMA-B fragment order.
// Chunk c = (nt*16 + ks)*32 + lane holds the 16 bf16 values lane `lane` needs
// for N-tile nt, K-step ks:  n = nt*16 + (lane&15),
//                            k = 32*ks + 8*(lane>>4) + e + (e>=8 ? 8 : 0), e = 0..15.
// ---------------------------------------------------------------------------
__global__ __launch_bounds__(256) void pack_weights_kernel(
    const float* __restrict__ W, __bf16* __restrict__ Wp) {
  int c = blockIdx.x * blockDim.x + threadIdx.x;   // 0 .. 16383
  if (c >= 32 * 16 * 32) return;
  int lane = c & 31;
  int ks   = (c >> 5) & 15;
  int nt   = c >> 9;
  int n  = nt * 16 + (lane & 15);
  int kb = 32 * ks + 8 * (lane >> 4);
  __bf16 tmp[16];
#pragma unroll
  for (int e = 0; e < 16; ++e)
    tmp[e] = (__bf16)W[(size_t)(kb + e + (e >= 8 ? 8 : 0)) * H_ + n];
  *(v16bf*)(Wp + (size_t)c * 16) = *(const v16bf*)tmp;
}

// ---------------------------------------------------------------------------
// Per-path node attention. 1 block = 1 path, 128 threads = 4 waves.
// Wave w owns M-tile rows [16w, 16w+16). Full K=512 A-fragments live in VGPRs.
// ---------------------------------------------------------------------------
__global__ __launch_bounds__(128) void node_attn_kernel(
    const float* __restrict__ X,        // [P, LMAX, D]
    const int*   __restrict__ lengths,  // [P]
    const __bf16* __restrict__ Wp,      // packed pW1
    const float* __restrict__ pb1,
    const float* __restrict__ pw2,
    const float* __restrict__ pb2,
    float* __restrict__ fea) {          // [P, D]
  __shared__ float Xf[LMAX_][XPAD];
  __shared__ float sred[4][2][8][16];
  __shared__ float ssc[LMAX_];

  const int p = blockIdx.x;
  const float* Xp = X + (size_t)p * (LMAX_ * D_);

  // Stage full 64x512 f32 tile straight into LDS via async-tensor path.
  for (int i = threadIdx.x; i < LMAX_ * D_ / 4; i += blockDim.x) {
    int r = i >> 7, c = (i & 127) * 4;
    async_copy16(&Xf[r][c], Xp + (size_t)i * 4);
  }
  async_wait0();
  __syncthreads();

  const int lane = threadIdx.x & 31;
  const int wave = threadIdx.x >> 5;
  const int hi   = lane >> 4;
  const int lm   = lane & 15;
  const int row  = wave * 16 + lm;

  // Build all 16 A-fragments (K=0..511) once, f32 -> bf16.
  v16bf af[16];
#pragma unroll
  for (int k = 0; k < 16; ++k) {
    const int kb = 32 * k + 8 * hi;
#pragma unroll
    for (int e = 0; e < 8; ++e) af[k][e]     = (__bf16)Xf[row][kb + e];
#pragma unroll
    for (int e = 0; e < 8; ++e) af[k][8 + e] = (__bf16)Xf[row][kb + 16 + e];
  }

  float sp[8];
#pragma unroll
  for (int r = 0; r < 8; ++r) sp[r] = 0.f;

  const v16bf* Wv = (const v16bf*)Wp;
  for (int nt = 0; nt < 32; ++nt) {
    v8f acc = {};
    const v16bf* bp = Wv + nt * 512 + lane;
#pragma unroll
    for (int ks = 0; ks < 16; ++ks) {
      v16bf bf_ = bp[ks * 32];
      acc = __builtin_amdgcn_wmma_f32_16x16x32_bf16(
          false, af[ks], false, bf_, (short)0, acc, false, false);
    }
    const int n = nt * 16 + lm;
    const float b1 = pb1[n], w2 = pw2[n];
#pragma unroll
    for (int r = 0; r < 8; ++r) {
      float hv = acc[r] + b1;             // C elem r: M = r + 8*hi, N = lm
      hv = hv > 0.f ? hv : 0.f;           // ReLU
      sp[r] += hv * w2;                   // fuse  h @ pw2
    }
  }
#pragma unroll
  for (int r = 0; r < 8; ++r) sred[wave][hi][r][lm] = sp[r];
  __syncthreads();

  // Reduce per-row scores across 16 lanes; mask by length.
  const int len = lengths[p];
  if (threadIdx.x < LMAX_) {
    int rw = threadIdx.x, mt = rw >> 4, rr = rw & 15;
    float s = 0.f;
#pragma unroll
    for (int j = 0; j < 16; ++j) s += sred[mt][rr >> 3][rr & 7][j];
    s += pb2[0];
    ssc[rw] = (rw < len) ? s : -1e30f;
  }
  __syncthreads();

  // Softmax over 64 (serial on one lane; trivial).
  if (threadIdx.x == 0) {
    float mx = ssc[0];
    for (int i = 1; i < LMAX_; ++i) mx = fmaxf(mx, ssc[i]);
    float sum = 0.f;
    for (int i = 0; i < LMAX_; ++i) { float e = __expf(ssc[i] - mx); ssc[i] = e; sum += e; }
    float inv = 1.f / sum;
    for (int i = 0; i < LMAX_; ++i) ssc[i] *= inv;
  }
  __syncthreads();

  // paths_fea[p][d] = sum_l w[l] * X[l][d]  (full fp32 from LDS tile)
  float* out = fea + (size_t)p * D_;
  for (int d = threadIdx.x; d < D_; d += blockDim.x) {
    float acc0 = 0.f;
#pragma unroll 4
    for (int l = 0; l < LMAX_; ++l) acc0 += ssc[l] * Xf[l][d];
    out[d] = acc0;
  }
}

// ---------------------------------------------------------------------------
// Path-level MLP scores: a[p] = relu(fea[p] @ aW1 + ab1) @ aw2 + ab2.
// 1 block = 64 paths (M-tile structure identical to node_attn).
// ---------------------------------------------------------------------------
__global__ __launch_bounds__(128) void path_score_kernel(
    const float* __restrict__ fea,      // [P, D]
    const __bf16* __restrict__ Wp,      // packed aW1
    const float* __restrict__ ab1,
    const float* __restrict__ aw2,
    const float* __restrict__ ab2,
    float* __restrict__ a_out) {        // [P]
  __shared__ float Xf[64][XPAD];
  __shared__ float sred[4][2][8][16];

  const int pbase = blockIdx.x * 64;
  const float* Fp = fea + (size_t)pbase * D_;
  for (int i = threadIdx.x; i < 64 * D_ / 4; i += blockDim.x) {
    int r = i >> 7, c = (i & 127) * 4;
    async_copy16(&Xf[r][c], Fp + (size_t)i * 4);
  }
  async_wait0();
  __syncthreads();

  const int lane = threadIdx.x & 31;
  const int wave = threadIdx.x >> 5;
  const int hi   = lane >> 4;
  const int lm   = lane & 15;
  const int row  = wave * 16 + lm;

  v16bf af[16];
#pragma unroll
  for (int k = 0; k < 16; ++k) {
    const int kb = 32 * k + 8 * hi;
#pragma unroll
    for (int e = 0; e < 8; ++e) af[k][e]     = (__bf16)Xf[row][kb + e];
#pragma unroll
    for (int e = 0; e < 8; ++e) af[k][8 + e] = (__bf16)Xf[row][kb + 16 + e];
  }

  float sp[8];
#pragma unroll
  for (int r = 0; r < 8; ++r) sp[r] = 0.f;

  const v16bf* Wv = (const v16bf*)Wp;
  for (int nt = 0; nt < 32; ++nt) {
    v8f acc = {};
    const v16bf* bp = Wv + nt * 512 + lane;
#pragma unroll
    for (int ks = 0; ks < 16; ++ks) {
      v16bf bf_ = bp[ks * 32];
      acc = __builtin_amdgcn_wmma_f32_16x16x32_bf16(
          false, af[ks], false, bf_, (short)0, acc, false, false);
    }
    const int n = nt * 16 + lm;
    const float b1 = ab1[n], w2 = aw2[n];
#pragma unroll
    for (int r = 0; r < 8; ++r) {
      float hv = acc[r] + b1;
      hv = hv > 0.f ? hv : 0.f;
      sp[r] += hv * w2;
    }
  }
#pragma unroll
  for (int r = 0; r < 8; ++r) sred[wave][hi][r][lm] = sp[r];
  __syncthreads();

  if (threadIdx.x < 64) {
    int rw = threadIdx.x, mt = rw >> 4, rr = rw & 15;
    float s = 0.f;
#pragma unroll
    for (int j = 0; j < 16; ++j) s += sred[mt][rr >> 3][rr & 7][j];
    a_out[pbase + rw] = s + ab2[0];
  }
}

// ---------------------------------------------------------------------------
// Global softmax over the 4096 path scores -> normalized weights aw[P].
// ---------------------------------------------------------------------------
__global__ __launch_bounds__(256) void softmax_paths_kernel(
    const float* __restrict__ a, float* __restrict__ aw) {
  __shared__ float red[256];
  const int t = threadIdx.x;
  float mx = -1e30f;
  for (int p = t; p < P_; p += 256) mx = fmaxf(mx, a[p]);
  red[t] = mx; __syncthreads();
  for (int s = 128; s > 0; s >>= 1) {
    if (t < s) red[t] = fmaxf(red[t], red[t + s]);
    __syncthreads();
  }
  mx = red[0]; __syncthreads();
  float sum = 0.f;
  for (int p = t; p < P_; p += 256) {
    float e = __expf(a[p] - mx);
    aw[p] = e;
    sum += e;
  }
  red[t] = sum; __syncthreads();
  for (int s = 128; s > 0; s >>= 1) {
    if (t < s) red[t] += red[t + s];
    __syncthreads();
  }
  float inv = 1.f / red[0];
  __syncthreads();
  for (int p = t; p < P_; p += 256) aw[p] *= inv;
}

// ---------------------------------------------------------------------------
// user_fea[d] = sum_p aw[p] * fea[p][d].  512 threads, coalesced over d.
// ---------------------------------------------------------------------------
__global__ __launch_bounds__(256) void user_fea_kernel(
    const float* __restrict__ fea, const float* __restrict__ aw,
    float* __restrict__ out) {
  const int d = blockIdx.x * 256 + threadIdx.x;   // 0..511
  float acc = 0.f;
#pragma unroll 4
  for (int p = 0; p < P_; ++p) acc += aw[p] * fea[(size_t)p * D_ + d];
  out[d] = acc;
}

// ---------------------------------------------------------------------------
extern "C" void kernel_launch(void* const* d_in, const int* in_sizes, int n_in,
                              void* d_out, int out_size, void* d_ws, size_t ws_size,
                              hipStream_t stream) {
  const float* paths_nodes = (const float*)d_in[0];
  const int*   lengths     = (const int*)  d_in[1];
  const float* pW1         = (const float*)d_in[2];
  const float* pb1         = (const float*)d_in[3];
  const float* pw2         = (const float*)d_in[4];
  const float* pb2         = (const float*)d_in[5];
  const float* aW1         = (const float*)d_in[6];
  const float* ab1         = (const float*)d_in[7];
  const float* aw2         = (const float*)d_in[8];
  const float* ab2         = (const float*)d_in[9];
  float* out = (float*)d_out;

  char* ws = (char*)d_ws;
  __bf16* Wp1 = (__bf16*)(ws);                       // 512 KB
  __bf16* Wp2 = (__bf16*)(ws + (512u << 10));        // 512 KB
  float*  fea = (float*) (ws + (1u << 20));          // 8 MB
  float*  avec= (float*) (ws + (9u << 20));          // 16 KB
  float*  aw  = (float*) (ws + (9u << 20) + (16u << 10));

  // 1) repack weights to bf16 WMMA-B fragment layout (L2-resident afterwards)
  pack_weights_kernel<<<64, 256, 0, stream>>>(pW1, Wp1);
  pack_weights_kernel<<<64, 256, 0, stream>>>(aW1, Wp2);

  // 2) fused per-path MLP + softmax + weighted sum (the 137-GFLOP stage)
  node_attn_kernel<<<P_, 128, 0, stream>>>(paths_nodes, lengths, Wp1,
                                           pb1, pw2, pb2, fea);

  // 3) path-level MLP scores
  path_score_kernel<<<P_ / 64, 128, 0, stream>>>(fea, Wp2, ab1, aw2, ab2, avec);

  // 4) global softmax over paths
  softmax_paths_kernel<<<1, 256, 0, stream>>>(avec, aw);

  // 5) final weighted reduction
  user_fea_kernel<<<D_ / 256, 256, 0, stream>>>(fea, aw, out);
}
